// NeuralSampler_v2_TransLSTMRes_57775900066403
// MI455X (gfx1250) — matrix operand order repaired and checked by reading, not compiled, not run on hardware
//
#include <hip/hip_runtime.h>

typedef _Float16 h16;
typedef __attribute__((ext_vector_type(16))) _Float16 v16h;
typedef __attribute__((ext_vector_type(8)))  _Float16 v8h;
typedef __attribute__((ext_vector_type(4)))  _Float16 h16x4;
typedef __attribute__((ext_vector_type(2)))  _Float16 h16x2;
typedef __attribute__((ext_vector_type(8)))  float    v8f;

constexpr int kB   = 16;
constexpr int kT   = 3000;
constexpr int kD   = 128;
constexpr int kH   = 2;     // heads
constexpr int kHD  = 64;    // head dim
constexpr int kNE  = 3;     // encoder layers
constexpr int kTOT = 750;
constexpr int kM   = kB * kT;              // 48000 rows
constexpr long kNEL = (long)kM * kD;       // 6,144,000 elems

constexpr long kEMB_OFF   = (long)kB * 2 * kTOT * kD;        // 3,072,000
constexpr long kLOSS_OFF  = kEMB_OFF + (long)kB * kTOT * kD; // 4,608,000
constexpr long kSCORE_OFF = kLOSS_OFF + 1;

__device__ __forceinline__ float sigm(float x) { return 1.f / (1.f + expf(-x)); }

__device__ __forceinline__ v16h mk16(v8h lo, v8h hi) {
  return __builtin_shufflevector(lo, hi, 0, 1, 2, 3, 4, 5, 6, 7, 8, 9, 10, 11, 12, 13, 14, 15);
}
// A fragment (16x32 f16): lane row fixed, one 32B run at half*16
__device__ __forceinline__ v16h ld_a(const h16* row, int half) {
  return mk16(*(const v8h*)(row + half * 16), *(const v8h*)(row + half * 16 + 8));
}
// B fragment (32x16 f16): lane col fixed, one 32B run at half*16
__device__ __forceinline__ v16h ld_b(const h16* col, int half) {
  return mk16(*(const v8h*)(col + half * 16), *(const v8h*)(col + half * 16 + 8));
}
// async 16B global -> LDS copy (ASYNCcnt-tracked, cdna5 CDNA5 ISA)
__device__ __forceinline__ void async_ld128(unsigned ldsAddr, const void* gaddr) {
  asm volatile("global_load_async_to_lds_b128 %0, %1, off" :: "v"(ldsAddr), "v"(gaddr) : "memory");
}

// ---------------- elementwise helpers (vectorized x4) ----------------
__global__ void k_cvt4(const float4* __restrict__ s, h16x4* __restrict__ d, long n4) {
  long i = (long)blockIdx.x * blockDim.x + threadIdx.x;
  if (i >= n4) return;
  float4 v = s[i];
  h16x4 o = {(h16)v.x, (h16)v.y, (h16)v.z, (h16)v.w};
  d[i] = o;
}
__global__ void k_vadd(const float* __restrict__ a, const float* __restrict__ b,
                       float* __restrict__ c, int n) {
  int i = blockIdx.x * blockDim.x + threadIdx.x;
  if (i < n) c[i] = a[i] + b[i];
}
__global__ void k_addpos(const float4* __restrict__ x, const float4* __restrict__ pos,
                         float4* __restrict__ h, h16x4* __restrict__ hh, long n4, long pn4) {
  long i = (long)blockIdx.x * blockDim.x + threadIdx.x;
  if (i >= n4) return;
  float4 xv = x[i], pv = pos[i % pn4];
  float4 v = make_float4(xv.x + pv.x, xv.y + pv.y, xv.z + pv.z, xv.w + pv.w);
  h[i] = v;
  h16x4 o = {(h16)v.x, (h16)v.y, (h16)v.z, (h16)v.w};
  hh[i] = o;
}
__global__ void k_addy(const float4* __restrict__ h, const float4* __restrict__ x,
                       h16x4* __restrict__ y, long n4) {
  long i = (long)blockIdx.x * blockDim.x + threadIdx.x;
  if (i >= n4) return;
  float4 a = h[i], b = x[i];
  h16x4 o = {(h16)(a.x + b.x), (h16)(a.y + b.y), (h16)(a.z + b.z), (h16)(a.w + b.w)};
  y[i] = o;
}
__global__ void k_copy4(const float4* __restrict__ s, float4* __restrict__ d, long n4) {
  long i = (long)blockIdx.x * blockDim.x + threadIdx.x;
  if (i < n4) d[i] = s[i];
}
// pack conv weight [NE][D][D][3] f32 -> [NE][3][D][D] f16
__global__ void k_packw(const float* __restrict__ w, h16* __restrict__ wp, long n) {
  long i = (long)blockIdx.x * blockDim.x + threadIdx.x;
  if (i >= n) return;
  long tap = i % 3, rem = i / 3;
  long kin = rem % kD, o = (rem / kD) % kD, L = rem / (kD * kD);
  wp[((L * 3 + tap) * kD + o) * kD + kin] = (h16)w[i];
}

// ---------------- WMMA GEMM: Y[M,N] = X[M,128](f16) @ W[N,128]^T(f16) + bias ----------------
// one wave -> 16x32 output tile; M must be multiple of 16, N of 32
__global__ __launch_bounds__(32) void k_gemm(const h16* __restrict__ X, const h16* __restrict__ W,
                                             const float* __restrict__ bias,
                                             float* __restrict__ Yf, h16* __restrict__ Yh,
                                             int N, int relu) {
  const int lane = threadIdx.x, half = lane >> 4, nl = lane & 15;
  const int m0 = blockIdx.x * 16, n0 = blockIdx.y * 32;
  const h16* xr = X + (size_t)(m0 + nl) * 128;
  const h16* w0 = W + (size_t)(n0 + nl) * 128;
  const h16* w1 = W + (size_t)(n0 + 16 + nl) * 128;
  v8f acc0 = {}, acc1 = {};
#pragma unroll
  for (int k0 = 0; k0 < 128; k0 += 32) {
    v16h a  = ld_a(xr + k0, half);
    v16h b0 = ld_b(w0 + k0, half);
    v16h b1 = ld_b(w1 + k0, half);
    acc0 = __builtin_amdgcn_wmma_f32_16x16x32_f16(false, a, false, b0, (short)0, acc0, false, false);
    acc1 = __builtin_amdgcn_wmma_f32_16x16x32_f16(false, a, false, b1, (short)0, acc1, false, false);
  }
  float bn0 = bias ? bias[n0 + nl] : 0.f;
  float bn1 = bias ? bias[n0 + 16 + nl] : 0.f;
#pragma unroll
  for (int r = 0; r < 8; ++r) {
    int m = m0 + r + half * 8;
    float v0 = acc0[r] + bn0, v1 = acc1[r] + bn1;
    if (relu) { v0 = fmaxf(v0, 0.f); v1 = fmaxf(v1, 0.f); }
    size_t idx = (size_t)m * N + n0 + nl;
    if (Yf) { Yf[idx] = v0; Yf[idx + 16] = v1; }
    if (Yh) { Yh[idx] = (h16)v0; Yh[idx + 16] = (h16)v1; }
  }
}

// ---------------- conv1d(k=3, SAME) via WMMA; Wp packed [3][128][128] ----------------
__global__ __launch_bounds__(32) void k_conv3(const h16* __restrict__ X, const h16* __restrict__ Wp,
                                              const float* __restrict__ bias,
                                              float* __restrict__ Yf, h16* __restrict__ Yh, int relu) {
  const int lane = threadIdx.x, half = lane >> 4, nl = lane & 15;
  const int m0 = blockIdx.x * 16, n0 = blockIdx.y * 32;
  const int row = m0 + nl;
  const int bidx = row / kT, t = row % kT;
  v8f acc0 = {}, acc1 = {};
  v8h z8 = {};
#pragma unroll
  for (int tap = 0; tap < 3; ++tap) {
    int tt = t + tap - 1;
    bool ok = ((unsigned)tt < (unsigned)kT);
    const h16* xr = X + ((size_t)bidx * kT + (ok ? tt : 0)) * kD;
    const h16* w0 = Wp + ((size_t)tap * kD + n0 + nl) * kD;
    const h16* w1 = w0 + (size_t)16 * kD;
#pragma unroll
    for (int k0 = 0; k0 < 128; k0 += 32) {
      v8h lo = *(const v8h*)(xr + k0 + half * 16);
      v8h hi = *(const v8h*)(xr + k0 + half * 16 + 8);
      v16h a  = mk16(ok ? lo : z8, ok ? hi : z8);
      v16h b0 = ld_b(w0 + k0, half);
      v16h b1 = ld_b(w1 + k0, half);
      acc0 = __builtin_amdgcn_wmma_f32_16x16x32_f16(false, a, false, b0, (short)0, acc0, false, false);
      acc1 = __builtin_amdgcn_wmma_f32_16x16x32_f16(false, a, false, b1, (short)0, acc1, false, false);
    }
  }
  float bn0 = bias[n0 + nl], bn1 = bias[n0 + 16 + nl];
#pragma unroll
  for (int r = 0; r < 8; ++r) {
    int m = m0 + r + half * 8;
    float v0 = acc0[r] + bn0, v1 = acc1[r] + bn1;
    if (relu) { v0 = fmaxf(v0, 0.f); v1 = fmaxf(v1, 0.f); }
    size_t idx = (size_t)m * kD + n0 + nl;
    if (Yf) { Yf[idx] = v0; Yf[idx + 16] = v1; }
    if (Yh) { Yh[idx] = (h16)v0; Yh[idx + 16] = (h16)v1; }
  }
}

// ---------------- flash attention: one wave per (b, head, 16-row q tile) ----------------
// K/V tiles double-buffered in LDS via global_load_async_to_lds_b128 (ASYNCcnt pipelining)
__global__ __launch_bounds__(32) void k_attn(const h16* __restrict__ Qm, const h16* __restrict__ Km,
                                             const h16* __restrict__ Vm, h16* __restrict__ Om) {
  const int lane = threadIdx.x;
  const int half = lane >> 4, nl = lane & 15;
  const int t0 = blockIdx.x * 16, hd = blockIdx.y, b = blockIdx.z;
  __shared__ h16 kt[2][32 * 64];
  __shared__ h16 vt[2][32 * 64];
  __shared__ h16 pt[16 * 32];

  v16h qa0, qa1;
  {
    int t = t0 + nl; if (t >= kT) t = kT - 1;           // clamp: OOB rows never stored
    const h16* qp = Qm + (((size_t)b * kT + t) * kH + hd) * kHD;
    qa0 = ld_a(qp, half);
    qa1 = ld_a(qp + 32, half);
  }
  float mrow[8], lrow[8];
  v8f z = {};
  v8f oa0 = z, oa1 = z, oa2 = z, oa3 = z;
#pragma unroll
  for (int r = 0; r < 8; ++r) { mrow[r] = -3.0e38f; lrow[r] = 0.f; }

  // stage one 32-key tile (one 128B row per lane): 16 async b128 copies
  auto stage = [&](int kb0, int buf) {
    int key = kb0 + lane; if (key >= kT) key = kT - 1;  // clamp: masked via -inf logits
    size_t src = (((size_t)b * kT + key) * kH + hd) * kHD;
    const char* kg = (const char*)(Km + src);
    const char* vg = (const char*)(Vm + src);
    unsigned kl = (unsigned)(unsigned long long)(&kt[buf][lane * 64]);
    unsigned vl = (unsigned)(unsigned long long)(&vt[buf][lane * 64]);
#pragma unroll
    for (int i = 0; i < 8; ++i) {
      async_ld128(kl + 16u * i, kg + 16 * i);
      async_ld128(vl + 16u * i, vg + 16 * i);
    }
  };
  stage(0, 0);

  for (int kb0 = 0; kb0 < kT; kb0 += 32) {
    const int buf = (kb0 >> 5) & 1;
    if (kb0 + 32 < kT) {
      stage(kb0 + 32, buf ^ 1);                          // prefetch next tile
      asm volatile("s_wait_asynccnt 16" ::: "memory");   // current tile's 16 copies done
    } else {
      asm volatile("s_wait_asynccnt 0" ::: "memory");
    }
    const h16* ktb = kt[buf];
    const h16* vtb = vt[buf];

    v8f s0 = z, s1 = z; // keys [kb0, kb0+16), [kb0+16, kb0+32)
#pragma unroll
    for (int c = 0; c < 2; ++c) {
      v16h b0 = ld_b(ktb + (size_t)nl * 64 + c * 32, half);
      v16h b1 = ld_b(ktb + (size_t)(16 + nl) * 64 + c * 32, half);
      const v16h qa = c ? qa1 : qa0;
      s0 = __builtin_amdgcn_wmma_f32_16x16x32_f16(false, qa, false, b0, (short)0, s0, false, false);
      s1 = __builtin_amdgcn_wmma_f32_16x16x32_f16(false, qa, false, b1, (short)0, s1, false, false);
    }
    bool oob0 = (kb0 + nl) >= kT;
    bool oob1 = (kb0 + 16 + nl) >= kT;
#pragma unroll
    for (int r = 0; r < 8; ++r) {
      s0[r] = oob0 ? -1e30f : s0[r] * 0.125f;
      s1[r] = oob1 ? -1e30f : s1[r] * 0.125f;
    }
#pragma unroll
    for (int r = 0; r < 8; ++r) {
      float v = fmaxf(s0[r], s1[r]);
      v = fmaxf(v, __shfl_xor(v, 1, 32));
      v = fmaxf(v, __shfl_xor(v, 2, 32));
      v = fmaxf(v, __shfl_xor(v, 4, 32));
      v = fmaxf(v, __shfl_xor(v, 8, 32));
      float mn = fmaxf(mrow[r], v);
      float alpha = expf(mrow[r] - mn);
      mrow[r] = mn;
      float p0 = expf(s0[r] - mn), p1 = expf(s1[r] - mn);
      s0[r] = p0; s1[r] = p1;
      float ps = p0 + p1;
      ps += __shfl_xor(ps, 1, 32); ps += __shfl_xor(ps, 2, 32);
      ps += __shfl_xor(ps, 4, 32); ps += __shfl_xor(ps, 8, 32);
      lrow[r] = lrow[r] * alpha + ps;
      oa0[r] *= alpha; oa1[r] *= alpha; oa2[r] *= alpha; oa3[r] *= alpha;
    }
    // P (16x32) to LDS, reload as A fragment
#pragma unroll
    for (int r = 0; r < 8; ++r) {
      pt[(r + half * 8) * 32 + nl]      = (h16)s0[r];
      pt[(r + half * 8) * 32 + 16 + nl] = (h16)s1[r];
    }
    asm volatile("s_wait_dscnt 0" ::: "memory");
    v16h pa = ld_a(pt + nl * 32, half);
#pragma unroll
    for (int c = 0; c < 4; ++c) {
      v16h vf;
#pragma unroll
      for (int e = 0; e < 16; ++e) {
        int kb = (half << 4) + e;
        vf[e] = vtb[(size_t)kb * 64 + c * 16 + nl];
      }
      if (c == 0) oa0 = __builtin_amdgcn_wmma_f32_16x16x32_f16(false, pa, false, vf, (short)0, oa0, false, false);
      if (c == 1) oa1 = __builtin_amdgcn_wmma_f32_16x16x32_f16(false, pa, false, vf, (short)0, oa1, false, false);
      if (c == 2) oa2 = __builtin_amdgcn_wmma_f32_16x16x32_f16(false, pa, false, vf, (short)0, oa2, false, false);
      if (c == 3) oa3 = __builtin_amdgcn_wmma_f32_16x16x32_f16(false, pa, false, vf, (short)0, oa3, false, false);
    }
  }
#pragma unroll
  for (int r = 0; r < 8; ++r) {
    int t = t0 + r + half * 8;
    if (t < kT) {
      size_t ob = (((size_t)b * kT + t) * kH + hd) * kHD;
      float inv = 1.f / lrow[r];
      Om[ob + 0 * 16 + nl] = (h16)(oa0[r] * inv);
      Om[ob + 1 * 16 + nl] = (h16)(oa1[r] * inv);
      Om[ob + 2 * 16 + nl] = (h16)(oa2[r] * inv);
      Om[ob + 3 * 16 + nl] = (h16)(oa3[r] * inv);
    }
  }
}

// ---------------- residual + LayerNorm: one wave per row ----------------
__global__ __launch_bounds__(32) void k_resln(float* __restrict__ H, const float* __restrict__ Dl,
                                              const float* __restrict__ g, const float* __restrict__ bt,
                                              h16* __restrict__ Hh) {
  int row = blockIdx.x, lane = threadIdx.x;
  size_t base = (size_t)row * kD + lane * 4;
  float4 hv = *(const float4*)(H + base);
  float4 dv = *(const float4*)(Dl + base);
  float v[4] = {hv.x + dv.x, hv.y + dv.y, hv.z + dv.z, hv.w + dv.w};
  float s1 = v[0] + v[1] + v[2] + v[3];
  float s2 = v[0] * v[0] + v[1] * v[1] + v[2] * v[2] + v[3] * v[3];
  for (int o = 1; o < 32; o <<= 1) { s1 += __shfl_xor(s1, o, 32); s2 += __shfl_xor(s2, o, 32); }
  float mean = s1 * (1.f / kD);
  float var  = s2 * (1.f / kD) - mean * mean;
  float inv  = rsqrtf(var + 1e-5f);
  float4 gv = *(const float4*)(g + lane * 4);
  float4 bv = *(const float4*)(bt + lane * 4);
  float y0 = (v[0] - mean) * inv * gv.x + bv.x;
  float y1 = (v[1] - mean) * inv * gv.y + bv.y;
  float y2 = (v[2] - mean) * inv * gv.z + bv.z;
  float y3 = (v[3] - mean) * inv * gv.w + bv.w;
  *(float4*)(H + base) = make_float4(y0, y1, y2, y3);
  h16x4 o4 = {(h16)y0, (h16)y1, (h16)y2, (h16)y3};
  *(h16x4*)(Hh + base) = o4;
}

// ---------------- LSTM recurrence: one wave per (batch, direction); units u=2*lane,2*lane+1 ----------------
__global__ __launch_bounds__(32) void k_lstm(const float* __restrict__ gin,   // [B*T*256]
                                             const float* __restrict__ whh,   // [256*64] f32
                                             h16* __restrict__ sout, int colOff, int rev) {
  int b = blockIdx.x, lane = threadIdx.x;
  __shared__ h16 wt[64 * 256];   // wt[j*256 + row] = whh[row*64 + j] (transposed)
  __shared__ float hs[64];
  for (int p = lane; p < 64 * 256; p += 32) {
    int j = p >> 8, row = p & 255;
    wt[p] = (h16)whh[row * 64 + j];
  }
  hs[lane] = 0.f; hs[lane + 32] = 0.f;
  __syncthreads();
  const int u = 2 * lane;
  float c0 = 0.f, c1 = 0.f;
  for (int step = 0; step < kT; ++step) {
    int t = rev ? (kT - 1 - step) : step;
    const float* g = gin + ((size_t)b * kT + t) * 256;
    float a0[4], a1[4];
#pragma unroll
    for (int q = 0; q < 4; ++q) {
      float2 gv = *(const float2*)(g + q * 64 + u);
      a0[q] = gv.x; a1[q] = gv.y;
    }
    for (int j = 0; j < 64; ++j) {
      float hj = hs[j];
#pragma unroll
      for (int q = 0; q < 4; ++q) {
        h16x2 wv = *(const h16x2*)(&wt[j * 256 + q * 64 + u]);
        a0[q] += (float)wv[0] * hj;
        a1[q] += (float)wv[1] * hj;
      }
    }
    float i0 = sigm(a0[0]), f0 = sigm(a0[1]), g0 = tanhf(a0[2]), o0 = sigm(a0[3]);
    float i1 = sigm(a1[0]), f1 = sigm(a1[1]), g1 = tanhf(a1[2]), o1 = sigm(a1[3]);
    c0 = f0 * c0 + i0 * g0;  c1 = f1 * c1 + i1 * g1;
    float h0 = o0 * tanhf(c0), h1 = o1 * tanhf(c1);
    hs[u] = h0; hs[u + 1] = h1;
    h16x2 hv = {(h16)h0, (h16)h1};
    *(h16x2*)(&sout[((size_t)b * kT + t) * kD + colOff + u]) = hv;
    asm volatile("s_wait_dscnt 0" ::: "memory");
  }
}

// ---------------- scoring / normalization / pooling ----------------
__global__ void k_score(const h16* __restrict__ s2, const float* __restrict__ lw,
                        const float* __restrict__ lb, float* __restrict__ score, int n) {
  int i = blockIdx.x * blockDim.x + threadIdx.x;
  if (i >= n) return;
  float a = lb[0];
  const v8h* row = (const v8h*)(s2 + (size_t)i * kD);
#pragma unroll
  for (int c = 0; c < 16; ++c) {
    v8h v = row[c];
#pragma unroll
    for (int j = 0; j < 8; ++j) a += (float)v[j] * lw[c * 8 + j];
  }
  score[i] = sigm(a);
}
__global__ void k_bsum(const float* __restrict__ score, float* __restrict__ sums,
                       float* __restrict__ sumsqs) {
  int b = blockIdx.x, tid = threadIdx.x;
  float s = 0, s2 = 0;
  for (int t = tid; t < kT; t += 256) { float v = score[(size_t)b * kT + t]; s += v; s2 += v * v; }
  __shared__ float sh[256], sh2[256];
  sh[tid] = s; sh2[tid] = s2; __syncthreads();
  for (int o = 128; o > 0; o >>= 1) {
    if (tid < o) { sh[tid] += sh[tid + o]; sh2[tid] += sh2[tid + o]; }
    __syncthreads();
  }
  if (tid == 0) { sums[b] = sh[0]; sumsqs[b] = sh2[0]; }
}
__global__ __launch_bounds__(32) void k_cumsum(const float* __restrict__ score,
                                               const float* __restrict__ sums,
                                               float* __restrict__ sc, float* __restrict__ cs) {
  int b = blockIdx.x, lane = threadIdx.x;
  float c = (float)kTOT / sums[b];
  float carry = 0.f;
  for (int base = 0; base < kT; base += 32) {
    int t = base + lane;
    float v = (t < kT) ? score[(size_t)b * kT + t] * c : 0.f;
    float orig = v;
    for (int o = 1; o < 32; o <<= 1) { float nv = __shfl_up(v, o, 32); if (lane >= o) v += nv; }
    if (t < kT) { sc[(size_t)b * kT + t] = orig; cs[(size_t)b * kT + t] = carry + v; }
    carry += __shfl(v, 31, 32);
  }
}
__global__ __launch_bounds__(32) void k_loss(const float* __restrict__ sums,
                                             const float* __restrict__ sumsqs,
                                             float* __restrict__ out) {
  int lane = threadIdx.x;
  float v = 0.f;
  if (lane < kB) {
    float s = sums[lane], s2 = sumsqs[lane];
    float c = (float)kTOT / s;
    float var = (s2 - s * s / (float)kT) / (float)(kT - 1);
    v = c * sqrtf(fmaxf(var, 0.f));
  }
  v += __shfl_xor(v, 1, 32); v += __shfl_xor(v, 2, 32);
  v += __shfl_xor(v, 4, 32); v += __shfl_xor(v, 8, 32);
  if (lane == 0) out[0] = v / (float)kB;
}
__global__ __launch_bounds__(128) void k_pool(const float* __restrict__ x, const float* __restrict__ pos,
                                              const float* __restrict__ sc, const float* __restrict__ cs,
                                              float* __restrict__ out) {
  int b = blockIdx.x / kTOT, l = blockIdx.x % kTOT, d = threadIdx.x;
  const float* csb = cs + (size_t)b * kT;
  float thr = (float)l;
  int lo, hi;
  { int a = 0, e = kT; while (a < e) { int m = (a + e) >> 1; if (csb[m] > thr) e = m; else a = m + 1; } lo = a; }
  { int a = lo, e = kT; while (a < e) { int m = (a + e) >> 1; if (csb[m] > thr + 1.0f) e = m; else a = m + 1; } hi = a; }
  float wsum = 0.f;
  for (int t = lo; t < hi; ++t) wsum += sc[(size_t)b * kT + t];
  float inv = 1.f / (wsum + 1e-8f);
  float fd = 0.f, ed = 0.f;
  for (int t = lo; t < hi; ++t) {
    float wn = sc[(size_t)b * kT + t] * inv;
    fd += wn * x[((size_t)b * kT + t) * kD + d];
    ed += wn * pos[(size_t)t * kD + d];
  }
  out[(((size_t)b * 2 + 0) * kTOT + l) * kD + d] = fd;
  out[(((size_t)b * 2 + 1) * kTOT + l) * kD + d] = ed;
  out[kEMB_OFF + ((size_t)b * kTOT + l) * kD + d] = ed;
}

// ---------------- host ----------------
extern "C" void kernel_launch(void* const* d_in, const int* in_sizes, int n_in,
                              void* d_out, int out_size, void* d_ws, size_t ws_size,
                              hipStream_t stream) {
  const float* x     = (const float*)d_in[0];
  const float* pos   = (const float*)d_in[1];
  const float* wq    = (const float*)d_in[2];
  const float* bq    = (const float*)d_in[3];
  const float* wk    = (const float*)d_in[4];
  const float* bk    = (const float*)d_in[5];
  const float* wv    = (const float*)d_in[6];
  const float* bv    = (const float*)d_in[7];
  const float* wo    = (const float*)d_in[8];
  const float* bo    = (const float*)d_in[9];
  const float* ln1g  = (const float*)d_in[10];
  const float* ln1b  = (const float*)d_in[11];
  const float* w1    = (const float*)d_in[12];
  const float* b1    = (const float*)d_in[13];
  const float* w2    = (const float*)d_in[14];
  const float* b2    = (const float*)d_in[15];
  const float* ln2g  = (const float*)d_in[16];
  const float* ln2b  = (const float*)d_in[17];
  const float* wih   = (const float*)d_in[18];
  const float* whh   = (const float*)d_in[19];
  const float* bih   = (const float*)d_in[20];
  const float* bhh   = (const float*)d_in[21];
  const float* linw  = (const float*)d_in[22];
  const float* linb  = (const float*)d_in[23];
  float* out = (float*)d_out;

  // workspace carve (~260 MB)
  char* p = (char*)d_ws;
  auto alloc = [&](size_t bytes) -> void* { void* r = (void*)p; p += (bytes + 255) & ~(size_t)255; return r; };
  float* h      = (float*)alloc(kNEL * 4);
  float* tmpf   = (float*)alloc(kNEL * 4);
  h16* hf16     = (h16*)alloc(kNEL * 2);
  h16* qf16     = (h16*)alloc(kNEL * 2);
  h16* kf16     = (h16*)alloc(kNEL * 2);
  h16* vf16     = (h16*)alloc(kNEL * 2);
  h16* of16     = (h16*)alloc(kNEL * 2);
  h16* t2f16    = (h16*)alloc(kNEL * 2);
  h16* s0f16    = (h16*)alloc(kNEL * 2);
  h16* s1f16    = (h16*)alloc(kNEL * 2);
  h16* s2f16    = (h16*)alloc(kNEL * 2);
  float* gin0   = (float*)alloc((size_t)kM * 256 * 4);
  float* gin1   = (float*)alloc((size_t)kM * 256 * 4);
  h16* wqf      = (h16*)alloc((size_t)kNE * kD * kD * 2);
  h16* wkf      = (h16*)alloc((size_t)kNE * kD * kD * 2);
  h16* wvf      = (h16*)alloc((size_t)kNE * kD * kD * 2);
  h16* wof      = (h16*)alloc((size_t)kNE * kD * kD * 2);
  h16* w1p      = (h16*)alloc((size_t)kNE * 3 * kD * kD * 2);
  h16* w2p      = (h16*)alloc((size_t)kNE * 3 * kD * kD * 2);
  h16* wihf     = (h16*)alloc((size_t)4 * 256 * kD * 2);
  float* biasc  = (float*)alloc(4 * 256 * 4);
  float* score  = (float*)alloc((size_t)kM * 4);
  float* scb    = (float*)alloc((size_t)kM * 4);
  float* csb    = (float*)alloc((size_t)kM * 4);
  float* sums   = (float*)alloc(kB * 4);
  float* sumsqs = (float*)alloc(kB * 4);

  auto cdiv = [](long a, long b) { return (int)((a + b - 1) / b); };

  // weight conversions / packing
  long nw = (long)kNE * kD * kD;
  k_cvt4<<<cdiv(nw / 4, 256), 256, 0, stream>>>((const float4*)wq, (h16x4*)wqf, nw / 4);
  k_cvt4<<<cdiv(nw / 4, 256), 256, 0, stream>>>((const float4*)wk, (h16x4*)wkf, nw / 4);
  k_cvt4<<<cdiv(nw / 4, 256), 256, 0, stream>>>((const float4*)wv, (h16x4*)wvf, nw / 4);
  k_cvt4<<<cdiv(nw / 4, 256), 256, 0, stream>>>((const float4*)wo, (h16x4*)wof, nw / 4);
  k_packw<<<cdiv(nw * 3, 256), 256, 0, stream>>>(w1, w1p, nw * 3);
  k_packw<<<cdiv(nw * 3, 256), 256, 0, stream>>>(w2, w2p, nw * 3);
  k_cvt4<<<cdiv(4L * 256 * kD / 4, 256), 256, 0, stream>>>((const float4*)wih, (h16x4*)wihf, 4L * 256 * kD / 4);
  k_vadd<<<4, 256, 0, stream>>>(bih, bhh, biasc, 4 * 256);

  // h = x + pos
  k_addpos<<<cdiv(kNEL / 4, 256), 256, 0, stream>>>((const float4*)x, (const float4*)pos,
                                                    (float4*)h, (h16x4*)hf16, kNEL / 4, (long)kT * kD / 4);

  dim3 gD(kM / 16, kD / 32), gG(kM / 16, 256 / 32), gC(kM / 16, kD / 32);
  dim3 ga((kT + 15) / 16, kH, kB);
  for (int i = 0; i < kNE; ++i) {
    size_t wOff = (size_t)i * kD * kD;
    k_gemm<<<gD, 32, 0, stream>>>(hf16, wqf + wOff, bq + i * kD, nullptr, qf16, kD, 0);
    k_gemm<<<gD, 32, 0, stream>>>(hf16, wkf + wOff, bk + i * kD, nullptr, kf16, kD, 0);
    k_gemm<<<gD, 32, 0, stream>>>(hf16, wvf + wOff, bv + i * kD, nullptr, vf16, kD, 0);
    k_attn<<<ga, 32, 0, stream>>>(qf16, kf16, vf16, of16);
    k_gemm<<<gD, 32, 0, stream>>>(of16, wof + wOff, bo + i * kD, tmpf, nullptr, kD, 0);
    k_resln<<<kM, 32, 0, stream>>>(h, tmpf, ln1g + i * kD, ln1b + i * kD, hf16);
    k_conv3<<<gC, 32, 0, stream>>>(hf16, w1p + wOff * 3, b1 + i * kD, nullptr, t2f16, 1);
    k_conv3<<<gC, 32, 0, stream>>>(t2f16, w2p + wOff * 3, b2 + i * kD, tmpf, nullptr, 0);
    k_resln<<<kM, 32, 0, stream>>>(h, tmpf, ln2g + i * kD, ln2b + i * kD, hf16);
  }
  // y = h + x -> LSTM input (f16)
  k_addy<<<cdiv(kNEL / 4, 256), 256, 0, stream>>>((const float4*)h, (const float4*)x, (h16x4*)s0f16, kNEL / 4);

  for (int l = 0; l < 2; ++l) {
    const h16* sin = l ? s1f16 : s0f16;
    h16* sout = l ? s2f16 : s1f16;
    k_gemm<<<gG, 32, 0, stream>>>(sin, wihf + (size_t)(l * 2 + 0) * 256 * kD,
                                  biasc + (l * 2 + 0) * 256, gin0, nullptr, 256, 0);
    k_gemm<<<gG, 32, 0, stream>>>(sin, wihf + (size_t)(l * 2 + 1) * 256 * kD,
                                  biasc + (l * 2 + 1) * 256, gin1, nullptr, 256, 0);
    k_lstm<<<kB, 32, 0, stream>>>(gin0, whh + (size_t)(l * 2 + 0) * 256 * 64, sout, 0, 0);
    k_lstm<<<kB, 32, 0, stream>>>(gin1, whh + (size_t)(l * 2 + 1) * 256 * 64, sout, 64, 1);
  }

  k_score<<<cdiv(kM, 256), 256, 0, stream>>>(s2f16, linw, linb, score, kM);
  k_bsum<<<kB, 256, 0, stream>>>(score, sums, sumsqs);
  k_cumsum<<<kB, 32, 0, stream>>>(score, sums, scb, csb);
  k_loss<<<1, 32, 0, stream>>>(sums, sumsqs, out + kLOSS_OFF);
  k_copy4<<<cdiv(kM / 4, 256), 256, 0, stream>>>((const float4*)score, (float4*)(out + kSCORE_OFF), kM / 4);
  k_pool<<<kB * kTOT, 128, 0, stream>>>(x, pos, scb, csb, out);
}
